// Rot3DEGNNLayer_1211180777855
// MI455X (gfx1250) — matrix operand order repaired and checked by reading, not compiled, not run on hardware
//
#include <hip/hip_runtime.h>

typedef __attribute__((ext_vector_type(16))) _Float16 v16h;
typedef __attribute__((ext_vector_type(8)))  float    v8f;

#define PACK4(V, base, f4)                         \
    do {                                           \
        V[(base) + 0] = (_Float16)(f4).x;          \
        V[(base) + 1] = (_Float16)(f4).y;          \
        V[(base) + 2] = (_Float16)(f4).z;          \
        V[(base) + 3] = (_Float16)(f4).w;          \
    } while (0)

__device__ __forceinline__ float4 ld4(const float* p) {
    return *reinterpret_cast<const float4*>(p);
}

// Rot3DE quaternion from relation embedding (matches reference do_trans math).
// Hardware trig: v_sin/v_cos take revolutions, so sin(pi*th) == v_sin(th*0.5).
// Axis normalization: 1/max(||t||,1e-12) == v_rsq(max(||t||^2,1e-24)).
__device__ __forceinline__ void rotq(float rx, float ry, float rz, float th,
                                     float& w, float& ux, float& uy, float& uz) {
    float rev = th * 0.5f;                       // (th * pi) / (2*pi)
    float s = __builtin_amdgcn_sinf(rev);
    float c = __builtin_amdgcn_cosf(rev);
    w = c;
    float tx = s * rx, ty = s * ry, tz = s * rz;
    float n2 = fmaf(tx, tx, fmaf(ty, ty, tz * tz));
    float inv = __builtin_amdgcn_rsqf(fmaxf(n2, 1e-24f));
    float ss = __builtin_amdgcn_sqrtf(fmaxf(1.0f - c * c, 0.0f));  // |sin|
    float k = ss * inv;
    ux = k * tx; uy = k * ty; uz = k * tz;
}

// One component of qv for segment seg (0:x_score, 1:y_score, 2:z_score, 3:w_score)
__device__ __forceinline__ float qv_dim(float ex, float ey, float ez,
                                        float rx, float ry, float rz, float th, int seg) {
    float w, ux, uy, uz;
    rotq(rx, ry, rz, th, w, ux, uy, uz);
    if (seg == 0) return w * ex + (uy * ez - uz * ey);
    if (seg == 1) return w * ey + (uz * ex - ux * ez);
    if (seg == 2) return w * ez + (ux * ey - uy * ex);
    return -(ux * ex + uy * ey + uz * ez);
}

// 4 consecutive qv components (features f..f+3, segment seg) via float4 loads.
__device__ __forceinline__ float4 qv4(const float* __restrict__ eh,
                                      const float* __restrict__ rr, int f, int seg) {
    float4 ex = ld4(eh + f), ey = ld4(eh + 64 + f), ez = ld4(eh + 128 + f);
    float4 rx = ld4(rr + f), ry = ld4(rr + 64 + f);
    float4 rz = ld4(rr + 128 + f), th = ld4(rr + 192 + f);
    float4 o;
    o.x = qv_dim(ex.x, ey.x, ez.x, rx.x, ry.x, rz.x, th.x, seg);
    o.y = qv_dim(ex.y, ey.y, ez.y, rx.y, ry.y, rz.y, th.y, seg);
    o.z = qv_dim(ex.z, ey.z, ez.z, rx.z, ry.z, rz.z, th.z, seg);
    o.w = qv_dim(ex.w, ey.w, ez.w, rx.w, ry.w, rz.w, th.w, seg);
    return o;
}

// Pass 0: zero output, init segment-max keys to -inf-key, denom to 0.
__global__ void k_init(float* __restrict__ out, int outN,
                       int* __restrict__ mkeys, float* __restrict__ denom, int nent) {
    int i = blockIdx.x * blockDim.x + threadIdx.x;
    if (i < outN) out[i] = 0.0f;
    if (i < nent) {
        mkeys[i] = (int)0x80000000;  // below any order-preserving float key
        denom[i] = 0.0f;
    }
}

// Pass 1: alpha = diag(K_tile x QV_tile^T)/16 via v_wmma_f32_16x16x32_f16.
// One wave == 16 edges; lane l owns edge (l&15); lane-half selects K range
// exactly as the CDNA5 16-bit A/B fragment layouts require (no shuffles).
// The K-chunk loop is deliberately NOT unrolled: one chunk's live set
// fits without spilling; full unroll spilled acc/A/B to scratch.
__global__ void __launch_bounds__(256)
k_alpha_wmma(const float* __restrict__ ent,
             const float* __restrict__ rel,
             const int* __restrict__ head,
             const int* __restrict__ relid,
             const int* __restrict__ tail,
             float* __restrict__ alphaE,
             int* __restrict__ mkeys,
             int E, int ntiles) {
    int lane = threadIdx.x & 31;
    int wave = blockIdx.x * (blockDim.x >> 5) + (threadIdx.x >> 5);
    if (wave >= ntiles) return;  // wave-uniform: EXEC stays all-ones for WMMA

    int m  = lane & 15;   // edge within tile (A row / B column)
    int hh = lane >> 4;   // lane half -> K sub-range

    int eidx = wave * 16 + m;
    int ec   = eidx < E ? eidx : (E - 1);  // clamp (select, no divergence)
    int h = head[ec], r = relid[ec], t = tail[ec];
    const float* eh = ent + (size_t)h * 192;
    const float* et = ent + (size_t)t * 192;  // key vector (first 192 dims; rest are 0)
    const float* rr = rel + (size_t)r * 256;

    v8f acc = {};
#pragma unroll 1
    for (int kc = 0; kc < 6; ++kc) {          // 6 x K=32 covers the 192 nonzero dims
        int seg = kc >> 1;                    // qv segment (x/y/z) -- wave-uniform
        int fb  = (kc & 1) << 5;              // feature base within segment

        // B fragment: B[k][n] = key[edge n][dim]; elems 0..7 -> K=8*hh+i, elems 8..15 -> K=16+8*hh+(i-8).
        const float* kp = et + seg * 64 + fb + 8 * hh;
        v16h bV;
        {
            float4 b0 = ld4(kp), b1 = ld4(kp + 4), b2 = ld4(kp + 16), b3 = ld4(kp + 20);
            PACK4(bV, 0, b0); PACK4(bV, 4, b1); PACK4(bV, 8, b2); PACK4(bV, 12, b3);
        }

        // A fragment: A[m][k] = qv[edge m][dim]; elems 0..7 -> K=8*hh+i,
        // elems 8..15 -> K=16+8*hh+(i-8)  (ISA 16-bit A layout).
        int fa = fb + 8 * hh;
        v16h aV;
        {
            float4 a0 = qv4(eh, rr, fa, seg);
            PACK4(aV, 0, a0);
        }
        {
            float4 a1 = qv4(eh, rr, fa + 4, seg);
            PACK4(aV, 4, a1);
        }
        {
            float4 a2 = qv4(eh, rr, fa + 16, seg);
            PACK4(aV, 8, a2);
        }
        {
            float4 a3 = qv4(eh, rr, fa + 20, seg);
            PACK4(aV, 12, a3);
        }

        acc = __builtin_amdgcn_wmma_f32_16x16x32_f16(
            false, aV, false, bV, (short)0, acc, false, false);
    }

    // Diagonal extraction: (e,e) lives in lane e (e<8, VGPR e) and
    // lane e+16 (e>=8, VGPR e-8). For both groups the diagonal edge equals
    // the lane's own edge m, so h/r/t loaded above are already correct.
    bool lo = lane < 8;
    bool hi = lane >= 24;
    int sel = lo ? lane : (lane - 24);
    float a = acc[0];
#pragma unroll
    for (int v = 1; v < 8; ++v)
        if (sel == v) a = acc[v];

    if ((lo || hi) && eidx < E) {
        float alpha = a * 0.0625f;  // 1/sqrt(256)
        alphaE[eidx] = alpha;
        // Order-preserving int key -> hardware atomic smax (deterministic).
        int key = __float_as_int(alpha);
        key = key >= 0 ? key : (key ^ 0x7fffffff);
        atomicMax(&mkeys[t], key);
    }
}

// Pass 2: e = exp(alpha - m[tail]) (in place), denom[tail] += e.
__global__ void k_expsum(const int* __restrict__ tail,
                         const int* __restrict__ mkeys,
                         float* __restrict__ alphaE,
                         float* __restrict__ denom, int E) {
    int e = blockIdx.x * blockDim.x + threadIdx.x;
    if (e >= E) return;
    int t = tail[e];
    int k = mkeys[t];
    k = k >= 0 ? k : (k ^ 0x7fffffff);
    float mval = __int_as_float(k);
    float ev = __expf(alphaE[e] - mval);
    alphaE[e] = ev;
    atomicAdd(&denom[t], ev);
}

// Pass 3: recompute qv per (edge, feature) and scatter qv*attn into out.
// 64 threads per edge; a wave covers half an edge (32 features) -> coalesced loads & atomics.
__global__ void k_scatter(const float* __restrict__ ent,
                          const float* __restrict__ rel,
                          const int* __restrict__ head,
                          const int* __restrict__ relid,
                          const int* __restrict__ tail,
                          const float* __restrict__ expE,
                          const float* __restrict__ denom,
                          float* __restrict__ out, int E) {
    int idx = blockIdx.x * blockDim.x + threadIdx.x;
    int e = idx >> 6;
    int f = idx & 63;
    if (e >= E) return;
    int h = head[e], r = relid[e], t = tail[e];
    const float* eh = ent + (size_t)h * 192;
    const float* rr = rel + (size_t)r * 256;
    float attn = expE[e] / denom[t];

    float ex = eh[f], ey = eh[64 + f], ez = eh[128 + f];
    float rx = rr[f], ry = rr[64 + f], rz = rr[128 + f], th = rr[192 + f];
    float w, ux, uy, uz;
    rotq(rx, ry, rz, th, w, ux, uy, uz);
    float xs = w * ex + (uy * ez - uz * ey);
    float ys = w * ey + (uz * ex - ux * ez);
    float zs = w * ez + (ux * ey - uy * ex);
    float ws = -(ux * ex + uy * ey + uz * ez);

    float* o = out + (size_t)t * 256 + f;
    atomicAdd(o,       xs * attn);
    atomicAdd(o + 64,  ys * attn);
    atomicAdd(o + 128, zs * attn);
    atomicAdd(o + 192, ws * attn);
}

extern "C" void kernel_launch(void* const* d_in, const int* in_sizes, int n_in,
                              void* d_out, int out_size, void* d_ws, size_t ws_size,
                              hipStream_t stream) {
    const float* entities  = (const float*)d_in[0];
    const float* relations = (const float*)d_in[1];
    const int*   ei        = (const int*)d_in[2];

    const int E    = in_sizes[2] / 3;
    const int NENT = in_sizes[0] / 192;  // 3 * HID

    const int* head  = ei;
    const int* relid = ei + E;
    const int* tail  = ei + 2 * E;

    float* out = (float*)d_out;

    char*  ws     = (char*)d_ws;
    int*   mkeys  = (int*)ws;                                   // NENT ints
    float* denom  = (float*)(ws + (size_t)NENT * 4);            // NENT floats
    float* alphaE = (float*)(ws + (size_t)NENT * 8);            // E floats (alpha -> exp, in place)

    // Pass 0: init
    int initThreads = out_size > NENT ? out_size : NENT;
    k_init<<<(initThreads + 255) / 256, 256, 0, stream>>>(out, out_size, mkeys, denom, NENT);

    // Pass 1: WMMA logits + segment max
    int ntiles  = (E + 15) / 16;
    int blocks1 = (ntiles + 7) / 8;  // 8 waves per 256-thread block
    k_alpha_wmma<<<blocks1, 256, 0, stream>>>(entities, relations, head, relid, tail,
                                              alphaE, mkeys, E, ntiles);

    // Pass 2: exp + denominator
    k_expsum<<<(E + 255) / 256, 256, 0, stream>>>(tail, mkeys, alphaE, denom, E);

    // Pass 3: weighted scatter
    long long total3 = (long long)E * 64;
    k_scatter<<<(int)((total3 + 255) / 256), 256, 0, stream>>>(entities, relations, head,
                                                               relid, tail, alphaE, denom,
                                                               out, E);
}
